// ScaledDotProductAttention_61125974557460
// MI455X (gfx1250) — matrix operand order repaired and checked by reading, not compiled, not run on hardware
//
#include <hip/hip_runtime.h>

// Relative-position attention, fused flash-style kernel for gfx1250 (MI455X).
// B=32, S=1024, D=64, CLIP=64. One workgroup (4 wave32) per (batch, 64-row q tile).
// Pre-pass converts K -> f16 row-major and V -> f16 transposed into d_ws; the hot
// loop double-buffers K/V/mask tiles via async global->LDS DMA (ASYNCcnt) so the
// DMA overlaps WMMA + softmax work. Softmax runs in the log2 domain (v_exp_f32
// without the per-exp v_mul that __expf would add).

#define BSZ   32
#define SEQ   1024
#define DIM   64
#define NKT   16          // k tiles of 64
#define RPAD  144         // 129 rel indices padded
#define KPAD  72          // 64 + 8 halves padding (16B-aligned 16-half runs)
#define MPAD  80          // mask tile row stride in bytes (16B-aligned chunks)

typedef _Float16 half_t;
typedef __attribute__((ext_vector_type(4)))  _Float16 v4h;
typedef __attribute__((ext_vector_type(8)))  _Float16 v8h;
typedef __attribute__((ext_vector_type(16))) _Float16 v16h;
typedef __attribute__((ext_vector_type(8)))  float    v8f;

// Pointee type expected by the async-to-LDS builtin (per hipcc diagnostic).
typedef int i32x4 __attribute__((vector_size(16)));

#if __has_builtin(__builtin_amdgcn_global_load_async_to_lds_b128) && \
    __has_builtin(__builtin_amdgcn_s_wait_asynccnt)
#define HAVE_ASYNC 1
#else
#define HAVE_ASYNC 0
#endif

static __device__ __forceinline__ void async_b128(const void* g, void* l) {
#if HAVE_ASYNC
  __builtin_amdgcn_global_load_async_to_lds_b128((i32x4*)g, (i32x4*)l, 0, 0);
#else
  *(uint4*)l = *(const uint4*)g;
#endif
}
static __device__ __forceinline__ void wait_async() {
#if HAVE_ASYNC
  __builtin_amdgcn_s_wait_asynccnt(0);
#endif
}

// B-fragment: 16 contiguous halves (K = 16*halfwave + e), 16B-aligned runs.
static __device__ __forceinline__ v16h ld16(const half_t* p) {
  v8h a = *(const v8h*)p;
  v8h c = *(const v8h*)(p + 8);
  v16h r;
#pragma unroll
  for (int i = 0; i < 8; ++i) { r[i] = a[i]; r[i + 8] = c[i]; }
  return r;
}

// A-fragment: two 8-element contiguous K runs per lane.
static __device__ __forceinline__ v16h ld8x2(const half_t* p0, const half_t* p1) {
  v8h a = *(const v8h*)p0;
  v8h c = *(const v8h*)p1;
  v16h r;
#pragma unroll
  for (int i = 0; i < 8; ++i) { r[i] = a[i]; r[i + 8] = c[i]; }
  return r;
}

// ---------------- pre-pass: K -> f16 [B][S][D], V -> f16 transposed [B][D][S] ----------------
__global__ __launch_bounds__(256)
void convert_kv_kernel(const float* __restrict__ k, const float* __restrict__ v,
                       half_t* __restrict__ kh, half_t* __restrict__ vt)
{
  const size_t i4   = (size_t)blockIdx.x * blockDim.x + threadIdx.x; // one float4 each
  const size_t base = i4 * 4;                                        // ((b*S + s)*D + d)
  float4 kf = ((const float4*)k)[i4];
  v4h khv = { (half_t)kf.x, (half_t)kf.y, (half_t)kf.z, (half_t)kf.w };
  *(v4h*)&kh[base] = khv;

  float4 vf = ((const float4*)v)[i4];
  const size_t d0 = base & (DIM - 1);
  const size_t s  = (base / DIM) & (SEQ - 1);
  const size_t b  = base / ((size_t)SEQ * DIM);
  half_t* vb = vt + b * (size_t)SEQ * DIM;
  vb[(d0 + 0) * SEQ + s] = (half_t)vf.x;
  vb[(d0 + 1) * SEQ + s] = (half_t)vf.y;
  vb[(d0 + 2) * SEQ + s] = (half_t)vf.z;
  vb[(d0 + 3) * SEQ + s] = (half_t)vf.w;
}

// ---------------- fused attention ----------------
template <bool PRECONV>
__global__ __launch_bounds__(128)
void relpos_attn_kernel(const float* __restrict__ q,
                        const float* __restrict__ k,
                        const float* __restrict__ v,
                        const unsigned char* __restrict__ mask,
                        const float* __restrict__ pos_table,
                        const half_t* __restrict__ kh,   // f16 [B][S][D]   (PRECONV)
                        const half_t* __restrict__ vth,  // f16 [B][D][S]   (PRECONV)
                        float* __restrict__ out)
{
  __shared__ half_t        K_lds [2][64 * KPAD];   // K tile, row-major [krow][d]
  __shared__ half_t        Vt_lds[2][64 * KPAD];   // V tile transposed [d][krow]
  __shared__ unsigned char M_lds [2][64 * MPAD];   // mask tile [qrow_local][kcol_local]
  __shared__ float         r_lds [64 * RPAD];      // rel logits [qrow_local][j]
  __shared__ half_t        P_lds [4 * 16 * KPAD];  // per-wave P scratch [row][kcol]

  const int tid  = threadIdx.x;
  const int wave = tid >> 5;
  const int lane = tid & 31;
  const int ln   = lane & 15;      // N / row-in-A index
  const int hh   = lane >> 4;      // half-wave select

  const int blk  = blockIdx.x;
  const int b    = blk >> 4;            // 16 q tiles per batch
  const int qblk = (blk & 15) * 64;

  // log2-domain scaling: softmax(exp(x*s)) == softmax2(exp2(x*s*log2e))
  const float l2scale = 0.125f * 1.44269504088896340736f;   // (1/sqrt(64)) * log2(e)
  const float NEG_INF = -__builtin_inff();

  // Stage K/V (f16) and mask tiles for k-tile kt2 into ping-pong buffer pp2.
  auto stage = [&](int kt2, int pp2) {
    if constexpr (PRECONV) {
      // 64 rows x 128B per matrix = 512 x 16B chunks; 4 chunks/thread/matrix.
#pragma unroll
      for (int cc = 0; cc < 4; ++cc) {
        const int chunk = tid * 4 + cc;        // 0..511
        const int row   = chunk >> 3;          // 0..63
        const int i8    = (chunk & 7) * 8;     // half offset within row
        async_b128(kh  + (((size_t)b * SEQ + kt2 * 64 + row) * DIM + i8),
                   &K_lds[pp2][row * KPAD + i8]);
        async_b128(vth + (((size_t)b * DIM + row) * SEQ + kt2 * 64 + i8),
                   &Vt_lds[pp2][row * KPAD + i8]);
      }
    } else {
      // fallback: convert f32 -> f16 in-kernel (2 threads per k/v row)
      const int row2 = tid >> 1;
      const int hsel = (tid & 1) * 32;
      const float* krp = k + ((size_t)b * SEQ + kt2 * 64 + row2) * DIM + hsel;
      const float* vrp = v + ((size_t)b * SEQ + kt2 * 64 + row2) * DIM + hsel;
#pragma unroll
      for (int i = 0; i < 8; ++i) {
        float4 kf = ((const float4*)krp)[i];
        v4h khv = { (half_t)kf.x, (half_t)kf.y, (half_t)kf.z, (half_t)kf.w };
        *(v4h*)&K_lds[pp2][row2 * KPAD + hsel + i * 4] = khv;
        float4 vf = ((const float4*)vrp)[i];
        Vt_lds[pp2][(hsel + i * 4 + 0) * KPAD + row2] = (half_t)vf.x;
        Vt_lds[pp2][(hsel + i * 4 + 1) * KPAD + row2] = (half_t)vf.y;
        Vt_lds[pp2][(hsel + i * 4 + 2) * KPAD + row2] = (half_t)vf.z;
        Vt_lds[pp2][(hsel + i * 4 + 3) * KPAD + row2] = (half_t)vf.w;
      }
    }
    // mask tile: 64 rows x 64B = 256 x 16B chunks; 2 chunks/thread
#pragma unroll
    for (int cc = 0; cc < 2; ++cc) {
      const int chunk = tid * 2 + cc;          // 0..255
      const int row   = chunk >> 2;            // 0..63
      const int i16   = (chunk & 3) * 16;      // byte offset within row
      async_b128(mask + ((size_t)b * SEQ + qblk + row) * SEQ + kt2 * 64 + i16,
                 &M_lds[pp2][row * MPAD + i16]);
    }
  };

  // ---------- Q A-fragments (16x32 f16), kept in registers whole kernel ----------
  const int qrowA = qblk + wave * 16 + ln;
  const float* qrp = q + ((size_t)b * SEQ + qrowA) * DIM;
  v16h qf[2];
#pragma unroll
  for (int c = 0; c < 2; ++c) {
    const float* p0 = qrp + c * 32 + hh * 16;       // K = 32c + 16h + e   (e=0..7)
    const float* p1 = qrp + c * 32 + hh * 16 + 8;   // K = 32c + 16h + 8 + e
#pragma unroll
    for (int e = 0; e < 8; ++e) {
      qf[c][e]     = (half_t)p0[e];
      qf[c][e + 8] = (half_t)p1[e];
    }
  }

  // ---------- r[qrow][j] = dot(q_row, pos_table[j]) via WMMA (9 j-tiles) ----------
#pragma unroll
  for (int jt = 0; jt < 9; ++jt) {
    int j = jt * 16 + ln; if (j > 128) j = 128;     // clamp OOB table rows
    const float* tp = pos_table + (size_t)j * DIM;
    v8f acc = {};
#pragma unroll
    for (int c = 0; c < 2; ++c) {
      const float* bp = tp + c * 32 + hh * 16;      // K(d) = 32c + 16h + e
      v16h bf;
#pragma unroll
      for (int e = 0; e < 16; ++e) bf[e] = (half_t)bp[e];
      acc = __builtin_amdgcn_wmma_f32_16x16x32_f16(false, qf[c], false, bf,
                                                   (short)0, acc, false, false);
    }
#pragma unroll
    for (int jj = 0; jj < 8; ++jj)
      r_lds[(wave * 16 + jj + 8 * hh) * RPAD + jt * 16 + ln] = acc[jj];
  }

  // prologue: stage tile 0 into buffer 0 (barrier also publishes r_lds)
  stage(0, 0);
  wait_async();
  __syncthreads();

  // fully-clipped rel-logit values per row (rel <= -64 -> idx 0; rel >= 64 -> idx 128)
  float rlo[8], rhi[8];
#pragma unroll
  for (int jj = 0; jj < 8; ++jj) {
    const int qrl = wave * 16 + jj + 8 * hh;
    rlo[jj] = r_lds[qrl * RPAD + 0];
    rhi[jj] = r_lds[qrl * RPAD + 128];
  }

  // ---------- online-softmax state (log2 domain) ----------
  float mrow[8], lrow[8];
#pragma unroll
  for (int jj = 0; jj < 8; ++jj) { mrow[jj] = NEG_INF; lrow[jj] = 0.f; }
  v8f O[4] = {};

  for (int kt = 0; kt < NKT; ++kt) {
    const int pp = kt & 1;
    if (kt + 1 < NKT) stage(kt + 1, pp ^ 1);   // DMA next tile while computing this one

    const half_t*        Kb = K_lds[pp];
    const half_t*        Vb = Vt_lds[pp];
    const unsigned char* Mb = M_lds[pp];

    // ---- S = Q Ktile^T : 4 n-tiles x 2 d-chunks of WMMA ----
    v8f sacc[4];
#pragma unroll
    for (int t = 0; t < 4; ++t) {
      v8f s = {};
#pragma unroll
      for (int c = 0; c < 2; ++c) {
        v16h bf = ld16(&Kb[(t * 16 + ln) * KPAD + c * 32 + hh * 16]);
        s = __builtin_amdgcn_wmma_f32_16x16x32_f16(false, qf[c], false, bf,
                                                   (short)0, s, false, false);
      }
      sacc[t] = s;
    }

    // ---- logits(log2) = (S + rel)*l2scale, mask, track per-row max ----
    // Block-uniform clip fast path: whole tile uses a constant table column when
    // |kt*64 - qblk| >= 127 (true for 13-14 of 16 iterations).
    const int  diff   = kt * 64 - qblk;
    const bool hiClip = diff >= 127;
    const bool loClip = diff <= -127;

    float pv[4][8];
    float tmax[8];
#pragma unroll
    for (int jj = 0; jj < 8; ++jj) tmax[jj] = NEG_INF;

    if (hiClip || loClip) {
#pragma unroll
      for (int t = 0; t < 4; ++t) {
#pragma unroll
        for (int jj = 0; jj < 8; ++jj) {
          const int qrl = wave * 16 + jj + 8 * hh;
          const float rv = hiClip ? rhi[jj] : rlo[jj];
          float lg = (sacc[t][jj] + rv) * l2scale;
          const unsigned char mz = Mb[qrl * MPAD + t * 16 + ln];
          lg = mz ? NEG_INF : lg;
          pv[t][jj] = lg;
          tmax[jj] = fmaxf(tmax[jj], lg);
        }
      }
    } else {
#pragma unroll
      for (int t = 0; t < 4; ++t) {
        const int kcol = kt * 64 + t * 16 + ln;
#pragma unroll
        for (int jj = 0; jj < 8; ++jj) {
          const int qrl = wave * 16 + jj + 8 * hh;
          const int qr  = qblk + qrl;
          int rel = kcol - qr;
          rel = rel < -64 ? -64 : (rel > 64 ? 64 : rel);
          float lg = (sacc[t][jj] + r_lds[qrl * RPAD + rel + 64]) * l2scale;
          const unsigned char mz = Mb[qrl * MPAD + t * 16 + ln];
          lg = mz ? NEG_INF : lg;
          pv[t][jj] = lg;
          tmax[jj] = fmaxf(tmax[jj], lg);
        }
      }
    }

#pragma unroll
    for (int jj = 0; jj < 8; ++jj) {                 // row max across 16 lanes
      float mx = tmax[jj];
      mx = fmaxf(mx, __shfl_xor(mx, 1, 16));
      mx = fmaxf(mx, __shfl_xor(mx, 2, 16));
      mx = fmaxf(mx, __shfl_xor(mx, 4, 16));
      mx = fmaxf(mx, __shfl_xor(mx, 8, 16));
      tmax[jj] = mx;
    }

    float alpha[8];
#pragma unroll
    for (int jj = 0; jj < 8; ++jj) {
      float mnew = fmaxf(mrow[jj], tmax[jj]);
      alpha[jj]  = (mnew == NEG_INF) ? 1.0f : exp2f(mrow[jj] - mnew);
      mrow[jj]   = mnew;
    }
    float rsum[8];
#pragma unroll
    for (int jj = 0; jj < 8; ++jj) rsum[jj] = 0.f;
#pragma unroll
    for (int t = 0; t < 4; ++t) {
#pragma unroll
      for (int jj = 0; jj < 8; ++jj) {
        float p = (mrow[jj] == NEG_INF) ? 0.f : exp2f(pv[t][jj] - mrow[jj]);
        pv[t][jj] = p;
        rsum[jj] += p;
      }
    }
#pragma unroll
    for (int jj = 0; jj < 8; ++jj) {                 // row sum across 16 lanes
      float ss = rsum[jj];
      ss += __shfl_xor(ss, 1, 16);
      ss += __shfl_xor(ss, 2, 16);
      ss += __shfl_xor(ss, 4, 16);
      ss += __shfl_xor(ss, 8, 16);
      lrow[jj] = lrow[jj] * alpha[jj] + ss;
    }
#pragma unroll
    for (int t = 0; t < 4; ++t) {
#pragma unroll
      for (int jj = 0; jj < 8; ++jj) O[t][jj] *= alpha[jj];
    }

    // ---- re-fragment P via per-wave LDS (same-wave DS ops are in-order) ----
    half_t* pw = &P_lds[wave * 16 * KPAD];
#pragma unroll
    for (int t = 0; t < 4; ++t) {
#pragma unroll
      for (int jj = 0; jj < 8; ++jj)
        pw[(jj + 8 * hh) * KPAD + t * 16 + ln] = (half_t)pv[t][jj];
    }

    // ---- O += P Vtile : 4 d-tiles x 2 k-chunks of WMMA ----
#pragma unroll
    for (int t = 0; t < 4; ++t) {
#pragma unroll
      for (int c = 0; c < 2; ++c) {
        v16h pa = ld8x2(&pw[ln * KPAD + c * 32 + 16 * hh],
                        &pw[ln * KPAD + c * 32 + 16 * hh + 8]);
        v16h bv = ld16(&Vb[(t * 16 + ln) * KPAD + c * 32 + hh * 16]);
        O[t] = __builtin_amdgcn_wmma_f32_16x16x32_f16(false, pa, false, bv,
                                                      (short)0, O[t], false, false);
      }
    }

    wait_async();        // next tile's DMA must have landed before we flip buffers
    __syncthreads();
  }

  // ---------- epilogue: O / l ----------
#pragma unroll
  for (int t = 0; t < 4; ++t) {
#pragma unroll
    for (int jj = 0; jj < 8; ++jj) {
      const int qrl = wave * 16 + jj + 8 * hh;
      const int qr  = qblk + qrl;
      out[((size_t)b * SEQ + qr) * DIM + t * 16 + ln] = O[t][jj] / lrow[jj];
    }
  }
}

extern "C" void kernel_launch(void* const* d_in, const int* in_sizes, int n_in,
                              void* d_out, int out_size, void* d_ws, size_t ws_size,
                              hipStream_t stream) {
  (void)in_sizes; (void)n_in; (void)out_size;
  const float*         q    = (const float*)d_in[0];
  const float*         k    = (const float*)d_in[1];
  const float*         v    = (const float*)d_in[2];
  const unsigned char* mask = (const unsigned char*)d_in[3];  // jnp.bool_ -> 1 byte/elem
  const float*         pt   = (const float*)d_in[4];
  float*               out  = (float*)d_out;

  const size_t elems = (size_t)BSZ * SEQ * DIM;               // 2M per tensor
  const size_t need  = 2 * elems * sizeof(half_t);            // 8 MB scratch

  dim3 grid(BSZ * (SEQ / 64));   // 512 workgroups
  dim3 block(128);               // 4 wave32

  if (ws_size >= need) {
    half_t* kh = (half_t*)d_ws;
    half_t* vt = kh + elems;
    convert_kv_kernel<<<dim3((unsigned)(elems / 4 / 256)), dim3(256), 0, stream>>>(k, v, kh, vt);
    relpos_attn_kernel<true><<<grid, block, 0, stream>>>(q, k, v, mask, pt, kh, vt, out);
  } else {
    relpos_attn_kernel<false><<<grid, block, 0, stream>>>(q, k, v, mask, pt, nullptr, nullptr, out);
  }
}
